// MFMHead_86973087744717
// MI455X (gfx1250) — matrix operand order repaired and checked by reading, not compiled, not run on hardware
//
#include <hip/hip_runtime.h>

typedef unsigned int  uint;
typedef unsigned short ushort;

typedef __attribute__((ext_vector_type(16))) __bf16 v16bf;
typedef __attribute__((ext_vector_type(8)))  float  v8f;

#define D        768
#define BN       8192
#define KSTEPS   24          // 768 / 32
#define HPAD     776         // 768 + 8 (bank-conflict pad)
#define NEGINF   (-10000.0f)

union V16BF { uint d[8]; v16bf v; };

// ---------- bf16 split helpers (round-to-nearest-even) ----------
__device__ __forceinline__ uint rne_bf16(float f) {
    uint u = __builtin_bit_cast(uint, f);
    return (u + 0x7FFFu + ((u >> 16) & 1u)) >> 16;
}
__device__ __forceinline__ float bf16_f32(uint h) {
    return __builtin_bit_cast(float, h << 16);
}

// ---------- fragment loaders ----------
// B fragment: 16 contiguous bf16 (32B, aligned) from a ushort plane
__device__ __forceinline__ v16bf load_b_frag(const ushort* p) {
    const uint4* q = (const uint4*)p;
    uint4 a = q[0], b = q[1];
    V16BF u;
    u.d[0]=a.x; u.d[1]=a.y; u.d[2]=a.z; u.d[3]=a.w;
    u.d[4]=b.x; u.d[5]=b.y; u.d[6]=b.z; u.d[7]=b.w;
    return u.v;
}
// A fragment from two 8-ushort runs (any bf16 plane)
__device__ __forceinline__ v16bf load_a_u16(const ushort* p0, const ushort* p1) {
    uint4 a = *(const uint4*)p0;
    uint4 b = *(const uint4*)p1;
    V16BF u;
    u.d[0]=a.x; u.d[1]=a.y; u.d[2]=a.z; u.d[3]=a.w;
    u.d[4]=b.x; u.d[5]=b.y; u.d[6]=b.z; u.d[7]=b.w;
    return u.v;
}
// A fragment from two 8-dword runs of packed (hi<<16|lo) values in LDS
__device__ __forceinline__ void load_a_pk(const uint* p0, const uint* p1,
                                          v16bf& ahi, v16bf& alo) {
    uint4 a = ((const uint4*)p0)[0], b = ((const uint4*)p0)[1];
    uint4 c = ((const uint4*)p1)[0], e = ((const uint4*)p1)[1];
    uint q[16] = {a.x,a.y,a.z,a.w, b.x,b.y,b.z,b.w,
                  c.x,c.y,c.z,c.w, e.x,e.y,e.z,e.w};
    V16BF H, L;
    #pragma unroll
    for (int i = 0; i < 8; ++i) {
        uint u0 = q[2*i], u1 = q[2*i+1];
        H.d[i] = (u0 >> 16)      | (u1 & 0xFFFF0000u);
        L.d[i] = (u0 & 0xFFFFu)  | (u1 << 16);
    }
    ahi = H.v;  alo = L.v;
}

__device__ __forceinline__ v8f wmma_bf(v16bf a, v16bf b, v8f c) {
    return __builtin_amdgcn_wmma_f32_16x16x32_bf16(
        false, a, false, b, (short)0, c, false, false);
}
// hi/lo compensated product: (Ah+Al)(Bh+Bl) ~ AhBh + AhBl + AlBh
__device__ __forceinline__ v8f wmma_split(v16bf ah, v16bf al,
                                          v16bf bh, v16bf bl, v8f c) {
    c = wmma_bf(ah, bh, c);
    c = wmma_bf(ah, bl, c);
    c = wmma_bf(al, bh, c);
    return c;
}

// ---------- kernel 0: f32 -> bf16 hi/lo planes ----------
__global__ void split_kernel(const float* __restrict__ src,
                             ushort* __restrict__ hi, ushort* __restrict__ lo, int n) {
    int i = blockIdx.x * blockDim.x + threadIdx.x;
    int stride = gridDim.x * blockDim.x;
    for (; i < n; i += stride) {
        float f = src[i];
        uint h = rne_bf16(f);
        hi[i] = (ushort)h;
        lo[i] = (ushort)rne_bf16(f - bf16_f32(h));
    }
}

// ---------- kernel 1: zero accumulators + count valid labels ----------
__global__ void init_valid_kernel(const int* __restrict__ idx, float* __restrict__ accum) {
    int tid = threadIdx.x;
    if (tid < 2) accum[tid] = 0.0f;     // [0]=numerator, [1]=valid count
    __syncthreads();
    float c = 0.0f;
    for (int i = tid; i < BN; i += 256) c += (idx[i] != -100) ? 1.0f : 0.0f;
    atomicAdd(&accum[1], c);
}

// ---------- kernel 2: fused MFM block (GEMM1+GELU -> LN -> GEMM2) ----------
// One block = one 16-row band. 8 waves; each wave owns 3 pairs of column tiles.
__global__ __launch_bounds__(256) void mfm_block_kernel(
    const ushort* __restrict__ xh,  const ushort* __restrict__ xl,
    const ushort* __restrict__ w1h, const ushort* __restrict__ w1l,
    const ushort* __restrict__ wdh, const ushort* __restrict__ wdl,
    const float*  __restrict__ b1,  const float* __restrict__ lng,
    const float*  __restrict__ lnb, const float* __restrict__ bdec,
    ushort* __restrict__ embh, ushort* __restrict__ embl) {

    __shared__ __align__(16) float hbuf[16 * HPAD];   // ~49.7 KB

    const int tid  = threadIdx.x;
    const int wave = tid >> 5, lane = tid & 31;
    const int half = lane >> 4, ln15 = lane & 15;
    const int rowbase = blockIdx.x * 16;

    // ---- Phase A: h = GELU(x @ W1^T + b1) -> LDS (f32) ----
    const ushort* axh = xh + (size_t)(rowbase + ln15) * D;
    const ushort* axl = xl + (size_t)(rowbase + ln15) * D;
    for (int t = 0; t < 3; ++t) {
        const int ncol0 = (wave * 6 + 2 * t) * 16 + ln15;
        const int ncol1 = ncol0 + 16;
        const ushort* bh0 = w1h + (size_t)ncol0 * D;
        const ushort* bl0 = w1l + (size_t)ncol0 * D;
        const ushort* bh1 = w1h + (size_t)ncol1 * D;
        const ushort* bl1 = w1l + (size_t)ncol1 * D;
        v8f acc0 = {}, acc1 = {};
        for (int kk = 0; kk < KSTEPS; ++kk) {
            const int k0 = kk * 32 + half * 16;
            const int kb = kk * 32 + half * 16;
            v16bf ah = load_a_u16(axh + k0, axh + k0 + 8);
            v16bf al = load_a_u16(axl + k0, axl + k0 + 8);
            acc0 = wmma_split(ah, al, load_b_frag(bh0 + kb), load_b_frag(bl0 + kb), acc0);
            acc1 = wmma_split(ah, al, load_b_frag(bh1 + kb), load_b_frag(bl1 + kb), acc1);
        }
        const float bias0 = b1[ncol0], bias1 = b1[ncol1];
        #pragma unroll
        for (int r = 0; r < 8; ++r) {
            float v0 = acc0[r] + bias0;
            float v1 = acc1[r] + bias1;
            float g0 = 0.5f * v0 * (1.0f + erff(v0 * 0.70710678118654752f));
            float g1 = 0.5f * v1 * (1.0f + erff(v1 * 0.70710678118654752f));
            hbuf[(r + 8 * half) * HPAD + ncol0] = g0;
            hbuf[(r + 8 * half) * HPAD + ncol1] = g1;
        }
    }
    __syncthreads();

    // ---- Phase B: LayerNorm per row; repack row as (hi<<16|lo) bf16 pair ----
    for (int rr = 0; rr < 2; ++rr) {
        const int row = wave * 2 + rr;
        float* hr = hbuf + row * HPAD;
        float s = 0.0f, ss = 0.0f;
        #pragma unroll
        for (int i = 0; i < 24; ++i) {
            float v = hr[lane + 32 * i];
            s += v; ss += v * v;
        }
        #pragma unroll
        for (int d = 16; d >= 1; d >>= 1) { s += __shfl_xor(s, d); ss += __shfl_xor(ss, d); }
        const float mu  = s * (1.0f / 768.0f);
        const float var = ss * (1.0f / 768.0f) - mu * mu;   // population var (jnp.var)
        const float rs  = rsqrtf(var + 1e-12f);
        uint* hp = (uint*)hr;
        #pragma unroll
        for (int i = 0; i < 24; ++i) {
            const int c = lane + 32 * i;
            float y = (hr[c] - mu) * rs * lng[c] + lnb[c];
            uint hi = rne_bf16(y);
            uint lo = rne_bf16(y - bf16_f32(hi));
            hp[c] = (hi << 16) | lo;
        }
    }
    __syncthreads();

    // ---- Phase C: emb = hn @ Wdec^T + b_dec -> global bf16 hi/lo planes ----
    const uint* hpk = (const uint*)hbuf + (size_t)ln15 * HPAD;
    for (int t = 0; t < 3; ++t) {
        const int ncol0 = (wave * 6 + 2 * t) * 16 + ln15;
        const int ncol1 = ncol0 + 16;
        const ushort* bh0 = wdh + (size_t)ncol0 * D;
        const ushort* bl0 = wdl + (size_t)ncol0 * D;
        const ushort* bh1 = wdh + (size_t)ncol1 * D;
        const ushort* bl1 = wdl + (size_t)ncol1 * D;
        v8f acc0 = {}, acc1 = {};
        for (int kk = 0; kk < KSTEPS; ++kk) {
            const int k0 = kk * 32 + half * 16;
            const int kb = kk * 32 + half * 16;
            v16bf ah, al;
            load_a_pk(hpk + k0, hpk + k0 + 8, ah, al);
            acc0 = wmma_split(ah, al, load_b_frag(bh0 + kb), load_b_frag(bl0 + kb), acc0);
            acc1 = wmma_split(ah, al, load_b_frag(bh1 + kb), load_b_frag(bl1 + kb), acc1);
        }
        const float bias0 = bdec[ncol0], bias1 = bdec[ncol1];
        #pragma unroll
        for (int r = 0; r < 8; ++r) {
            float v0 = acc0[r] + bias0;
            float v1 = acc1[r] + bias1;
            const size_t o0 = (size_t)(rowbase + r + 8 * half) * D + ncol0;
            const size_t o1 = o0 + 16;
            uint h0 = rne_bf16(v0), h1 = rne_bf16(v1);
            embh[o0] = (ushort)h0;
            embl[o0] = (ushort)rne_bf16(v0 - bf16_f32(h0));
            embh[o1] = (ushort)h1;
            embl[o1] = (ushort)rne_bf16(v1 - bf16_f32(h1));
        }
    }
}

// ---------- kernel 3: streaming logits + online masked log-softmax ----------
// One block = one 16-row band vs all 8192 columns; never materializes logits.
// Each wave processes tile PAIRS: shared A fragment, 2 accumulators, merged
// online-softmax update.
__global__ __launch_bounds__(256) void contrast_kernel(
    const ushort* __restrict__ embh, const ushort* __restrict__ embl,
    const ushort* __restrict__ labh, const ushort* __restrict__ labl,
    const float*  __restrict__ mask, const int* __restrict__ labidx,
    float* __restrict__ accum) {

    __shared__ float red_m[8][16], red_s[8][16], red_d[8][16];

    const int tid  = threadIdx.x;
    const int wave = tid >> 5, lane = tid & 31;
    const int half = lane >> 4, ln15 = lane & 15;
    const int rowbase = blockIdx.x * 16;

    float mrow[8], m_run[8], s_run[8], d_run[8];
    #pragma unroll
    for (int r = 0; r < 8; ++r) {
        mrow[r]  = mask[rowbase + r + 8 * half];
        m_run[r] = -1.0e30f;
        s_run[r] = 0.0f;
        d_run[r] = 0.0f;
    }

    const ushort* ahr = embh + (size_t)(rowbase + ln15) * D;
    const ushort* alr = embl + (size_t)(rowbase + ln15) * D;

    for (int t = 0; t < 32; ++t) {        // 512 column tiles / (8 waves * 2)
        const int jt0   = t * 16 + wave * 2;   // tile pair jt0, jt0+1
        const int colg0 = jt0 * 16 + ln15;
        const int colg1 = colg0 + 16;
        const float mcol0 = mask[colg0];
        const float mcol1 = mask[colg1];
        const ushort* bh0 = labh + (size_t)colg0 * D;
        const ushort* bl0 = labl + (size_t)colg0 * D;
        const ushort* bh1 = labh + (size_t)colg1 * D;
        const ushort* bl1 = labl + (size_t)colg1 * D;

        // keep next tile-pair's label stream warm (global_prefetch_b8)
        if (t < 31) {
            __builtin_prefetch(bh0 + 256 * D, 0, 3);
            __builtin_prefetch(bl0 + 256 * D, 0, 3);
        }

        v8f acc0 = {}, acc1 = {};
        for (int kk = 0; kk < KSTEPS; ++kk) {
            const int k0 = kk * 32 + half * 16;
            const int kb = kk * 32 + half * 16;
            v16bf ah = load_a_u16(ahr + k0, ahr + k0 + 8);
            v16bf al = load_a_u16(alr + k0, alr + k0 + 8);
            acc0 = wmma_split(ah, al, load_b_frag(bh0 + kb), load_b_frag(bl0 + kb), acc0);
            acc1 = wmma_split(ah, al, load_b_frag(bh1 + kb), load_b_frag(bl1 + kb), acc1);
        }

        #pragma unroll
        for (int r = 0; r < 8; ++r) {
            const int rowg = rowbase + r + 8 * half;
            float xv0 = acc0[r] + (1.0f - mrow[r] * mcol0) * NEGINF;
            float xv1 = acc1[r] + (1.0f - mrow[r] * mcol1) * NEGINF;
            // per-row tile-pair max across the 16 lanes of this half
            float tmax = fmaxf(xv0, xv1);
            #pragma unroll
            for (int d = 1; d < 16; d <<= 1) tmax = fmaxf(tmax, __shfl_xor(tmax, d));
            const float newm = fmaxf(m_run[r], tmax);
            float p = __expf(xv0 - newm) + __expf(xv1 - newm);
            #pragma unroll
            for (int d = 1; d < 16; d <<= 1) p += __shfl_xor(p, d);
            s_run[r] = s_run[r] * __expf(m_run[r] - newm) + p;
            m_run[r] = newm;
            if (colg0 == rowg) d_run[r] += xv0;   // masked diagonal logit
            if (colg1 == rowg) d_run[r] += xv1;
        }
    }

    #pragma unroll
    for (int r = 0; r < 8; ++r) {
        float ds_ = d_run[r];
        #pragma unroll
        for (int d = 1; d < 16; d <<= 1) ds_ += __shfl_xor(ds_, d);
        if (ln15 == 0) {
            red_m[wave][r + 8 * half] = m_run[r];
            red_s[wave][r + 8 * half] = s_run[r];
            red_d[wave][r + 8 * half] = ds_;
        }
    }
    __syncthreads();

    if (tid < 16) {
        float M = -1.0e30f;
        #pragma unroll
        for (int w = 0; w < 8; ++w) M = fmaxf(M, red_m[w][tid]);
        float S = 0.0f, Dg = 0.0f;
        #pragma unroll
        for (int w = 0; w < 8; ++w) {
            S  += red_s[w][tid] * __expf(red_m[w][tid] - M);
            Dg += red_d[w][tid];
        }
        const int rowg = rowbase + tid;
        if (labidx[rowg] != -100) {
            const float logpt = Dg - M - logf(S);
            atomicAdd(&accum[0], -logpt);
        }
    }
}

// ---------- kernel 4: loss = numerator / valid count ----------
__global__ void finalize_kernel(const float* __restrict__ accum, float* __restrict__ out) {
    out[0] = accum[0] / accum[1];
}

extern "C" void kernel_launch(void* const* d_in, const int* in_sizes, int n_in,
                              void* d_out, int out_size, void* d_ws, size_t ws_size,
                              hipStream_t stream) {
    const float* x      = (const float*)d_in[0];
    const float* labels = (const float*)d_in[1];
    const int*   labidx = (const int*)d_in[2];
    const float* mask   = (const float*)d_in[3];
    const float* W1     = (const float*)d_in[4];
    const float* b1     = (const float*)d_in[5];
    const float* lng    = (const float*)d_in[6];
    const float* lnb    = (const float*)d_in[7];
    const float* Wd     = (const float*)d_in[8];
    const float* bdec   = (const float*)d_in[9];
    float* out = (float*)d_out;

    char* ws = (char*)d_ws;
    size_t off = 0;
    auto alloc = [&](size_t bytes) -> void* {
        void* p = ws + off;
        off += (bytes + 255) & ~(size_t)255;
        return p;
    };
    ushort* labh = (ushort*)alloc((size_t)BN * D * 2);
    ushort* labl = (ushort*)alloc((size_t)BN * D * 2);
    ushort* embh = (ushort*)alloc((size_t)BN * D * 2);
    ushort* embl = (ushort*)alloc((size_t)BN * D * 2);
    ushort* xh   = (ushort*)alloc((size_t)BN * D * 2);
    ushort* xl   = (ushort*)alloc((size_t)BN * D * 2);
    ushort* w1h  = (ushort*)alloc((size_t)D * D * 2);
    ushort* w1l  = (ushort*)alloc((size_t)D * D * 2);
    ushort* wdh  = (ushort*)alloc((size_t)D * D * 2);
    ushort* wdl  = (ushort*)alloc((size_t)D * D * 2);
    float*  accum = (float*)alloc(256);

    split_kernel<<<2048, 256, 0, stream>>>(labels, labh, labl, BN * D);
    split_kernel<<<2048, 256, 0, stream>>>(x, xh, xl, BN * D);
    split_kernel<<<512,  256, 0, stream>>>(W1, w1h, w1l, D * D);
    split_kernel<<<512,  256, 0, stream>>>(Wd, wdh, wdl, D * D);
    init_valid_kernel<<<1, 256, 0, stream>>>(labidx, accum);
    mfm_block_kernel<<<BN / 16, 256, 0, stream>>>(xh, xl, w1h, w1l, wdh, wdl,
                                                  b1, lng, lnb, bdec, embh, embl);
    contrast_kernel<<<BN / 16, 256, 0, stream>>>(embh, embl, labh, labl,
                                                 mask, labidx, accum);
    finalize_kernel<<<1, 1, 0, stream>>>(accum, out);
}
